// SelfAttVisPosTrans_62672162783794
// MI455X (gfx1250) — matrix operand order repaired and checked by reading, not compiled, not run on hardware
//
#include <hip/hip_runtime.h>
#include <hip/hip_bf16.h>

typedef _Float16 half_t;
typedef __attribute__((ext_vector_type(16))) _Float16 v16h;
typedef __attribute__((ext_vector_type(8)))  _Float16 v8h;
typedef __attribute__((ext_vector_type(8)))  float    v8f;
typedef __attribute__((ext_vector_type(4)))  unsigned int u32x4;
typedef __attribute__((ext_vector_type(8)))  unsigned int u32x8;

#define B_  4
#define C_  256
#define H_  48
#define W_  48
#define N_  2304      /* H_*W_ */
#define M_  4
#define HD_ 64
#define F_  256
#define RR_ 7

// ---------------- fragment loaders (wave32 WMMA operands; this code's contiguous k order) ----------------
// A 16x32 f16 (MxK): lanes0-15: halves0-15=K0..15 ;
//                    lanes16-31: halves0-15=K16..31 ; M=lane&15
__device__ inline v16h load_a_frag(const half_t* __restrict__ base, int row_base,
                                   int ld, int k_base, int lane) {
    int m = lane & 15, hi = lane >> 4;
    const half_t* p = base + (size_t)(row_base + m) * ld + k_base + hi * 16;
    return *(const v16h*)p;
}

// B 32x16 f16 (KxN): lanes0-15 hold K=kbase..+15, lanes16-31 K=kbase+16..+31 ; N=lane&15
// source stored row-major as (n, k) with leading dim ld
__device__ inline v16h load_b_frag(const half_t* __restrict__ base, int col_base,
                                   int ld, int k_base, int lane) {
    int n = lane & 15, hi = lane >> 4;
    const half_t* p = base + (size_t)(col_base + n) * ld + k_base + hi * 16;
    return *(const v16h*)p;
}

__device__ inline v8f wmma_f16(v16h a, v16h b, v8f c) {
    return __builtin_amdgcn_wmma_f32_16x16x32_f16(false, a, false, b,
                                                  (short)0, c, false, false);
}

__device__ inline float redsum32(float v) {
#pragma unroll
    for (int m = 1; m <= 16; m <<= 1) v += __shfl_xor(v, m, 32);
    return v;
}
__device__ inline float redsum16(float v) {
#pragma unroll
    for (int m = 1; m <= 8; m <<= 1) v += __shfl_xor(v, m, 32);
    return v;
}
__device__ inline float redmax16(float v) {
#pragma unroll
    for (int m = 1; m <= 8; m <<= 1) v = fmaxf(v, __shfl_xor(v, m, 32));
    return v;
}

// ---------------- f32 -> f16 weight conversion ----------------
__global__ void f2h_kernel(const float* __restrict__ in, half_t* __restrict__ out, int n) {
    int i = blockIdx.x * 256 + threadIdx.x;
    if (i < n) out[i] = (half_t)in[i];
}

// ---------------- LayerNorm over channels: x(B,C,N) -> feat16 (B*N, C) ----------------
__global__ void ln_feat_kernel(const float* __restrict__ x, half_t* __restrict__ feat) {
    int lane = threadIdx.x;
    int row  = blockIdx.x * blockDim.y + threadIdx.y;      // b*N + n
    int b = row / N_, n = row % N_;
    const float* xp = x + (size_t)b * C_ * N_ + n;
    float v[8];
    float s = 0.f;
#pragma unroll
    for (int i = 0; i < 8; ++i) { v[i] = xp[(size_t)(lane + 32 * i) * N_]; s += v[i]; }
    s = redsum32(s);
    float mu = s * (1.f / C_);
    float q = 0.f;
#pragma unroll
    for (int i = 0; i < 8; ++i) { float d = v[i] - mu; q += d * d; }
    q = redsum32(q);
    float inv = rsqrtf(q * (1.f / C_) + 1e-12f);
    half_t* fp = feat + (size_t)row * C_;
#pragma unroll
    for (int i = 0; i < 8; ++i) fp[lane + 32 * i] = (half_t)((v[i] - mu) * inv);
}

// ---------------- fused QKV projection GEMM (rows = B*N, cols = 1536) ----------------
__global__ __launch_bounds__(128)
void qkv_gemm_kernel(const half_t* __restrict__ feat,
                     const half_t* __restrict__ wq, const half_t* __restrict__ wk,
                     const half_t* __restrict__ wv,
                     const float* __restrict__ bq, const float* __restrict__ bk,
                     half_t* __restrict__ q16, half_t* __restrict__ k16,
                     half_t* __restrict__ vt16) {
    int lane = threadIdx.x, wid = threadIdx.y;
    int rowbase = blockIdx.x * 16;
    int co = blockIdx.y * 64 + wid * 16;          // 0..1535
    const half_t* wsel; int cl;
    if      (co < 256) { wsel = wq; cl = co; }
    else if (co < 512) { wsel = wk; cl = co - 256; }
    else               { wsel = wv; cl = co - 512; }
    v8f acc = {};
    for (int k = 0; k < C_; k += 32) {
        v16h a = load_a_frag(feat, rowbase, C_, k, lane);
        v16h bf = load_b_frag(wsel, cl, C_, k, lane);
        acc = wmma_f16(a, bf, acc);
    }
    int n16 = lane & 15, hi = lane >> 4;
    int colg = co + n16;
    float bias = 0.f;
    if (co < 256) bias = bq[colg];
    else if (co < 512) bias = bk[colg - 256];
#pragma unroll
    for (int j = 0; j < 8; ++j) {
        int row = rowbase + j + hi * 8;
        int bb = row / N_, n = row % N_;
        float val = acc[j] + bias;
        if (co < 256) {
            int m = colg >> 6, hd = colg & 63;
            q16[(((size_t)(bb * M_ + m) * N_) + n) * HD_ + hd] = (half_t)val;
        } else if (co < 512) {
            int o = colg - 256; int m = o >> 6, hd = o & 63;
            k16[(((size_t)(bb * M_ + m) * N_) + n) * HD_ + hd] = (half_t)val;
        } else {
            int o = colg - 512; int m = o >> 8, f = o & 255;
            vt16[(((size_t)(bb * M_ + m) * F_) + f) * N_ + n] = (half_t)val;
        }
    }
}

// ---------------- flash attention with double-buffered DMA tile staging ----------------
// Block = 4 waves = 64 query rows of one (b,m). Per 32-key step:
//   * wave0 issues a TDM tensor_load_to_lds for the V tile (256 f-rows x 32 keys,
//     row stride N_) -- 16KB strided 2D DMA
//   * all 128 threads async-copy the contiguous K tile (32 keys x 64 hd = 4KB)
//     via global_load_async_to_lds_b128
// Tiles are double-buffered: step i+1's DMA is issued before step i's compute and
// waited with s_wait_asynccnt 2 / s_wait_tensorcnt 1 (per-wave in-order completion),
// hiding the L2 round-trip behind the 20-WMMA burst.
__global__ __launch_bounds__(128)
void attn_kernel(const half_t* __restrict__ q16, const half_t* __restrict__ k16,
                 const half_t* __restrict__ vt16, const float* __restrict__ ptab,
                 half_t* __restrict__ fusion16) {
    __shared__ __align__(64) float  tbl[240];
    __shared__ __align__(64) half_t kbuf[2][32 * HD_];   // 2x4KB : key-major, ld=64
    __shared__ __align__(64) half_t vbuf[2][F_ * 32];    // 2x16KB: f-major, ld=32
    __shared__ __align__(64) half_t pbuf[4][16 * 32];    // 4KB   : per-wave P tiles

    int lane = threadIdx.x, wid = threadIdx.y;
    int t = wid * 32 + lane;

    // async load of position-bias table (15x15) into LDS
    for (int i = t; i < (2 * RR_ + 1) * (2 * RR_ + 1); i += 128) {
        unsigned la = (unsigned)(unsigned long long)(&tbl[i]);
        asm volatile("global_load_async_to_lds_b32 %0, %1, %2"
                     :: "v"(la), "v"(i * 4), "s"(ptab) : "memory");
    }
    asm volatile("s_wait_asynccnt 0" ::: "memory");

    int m = blockIdx.y, b = blockIdx.z;
    int qbase = (blockIdx.x * 4 + wid) * 16;
    size_t bm = (size_t)(b * M_ + m);
    const half_t* qp = q16 + bm * N_ * HD_;
    const half_t* kp = k16 + bm * N_ * HD_;
    const half_t* vp = vt16 + bm * F_ * (size_t)N_;
    half_t* pb = pbuf[wid];
    int n16 = lane & 15, hi = lane >> 4;

    // uniform pieces of the TDM descriptor (D#), built once
    unsigned vb_lds0 = __builtin_amdgcn_readfirstlane((unsigned)(unsigned long long)&vbuf[0][0]);
    unsigned vb_lds1 = __builtin_amdgcn_readfirstlane((unsigned)(unsigned long long)&vbuf[1][0]);
    unsigned long long vp_addr = (unsigned long long)vp;
    u32x8 g1;
    g1[0] = 0x00010000u;                 // data_size = 1 -> 2 bytes/elem
    g1[1] = ((unsigned)N_ & 0xFFFFu) << 16;   // tensor_dim0[15:0] (=2304)
    g1[2] = ((unsigned)F_) << 16;        // tensor_dim0 hi=0 | tensor_dim1[15:0]=256
    g1[3] = 32u << 16;                   // tensor_dim1 hi=0 | tile_dim0=32
    g1[4] = (unsigned)F_;                // tile_dim1=256 | tile_dim2=0
    g1[5] = (unsigned)N_;                // tensor_dim0_stride[31:0]=2304
    g1[6] = 0u;                          // stride hi | dim1_stride lo
    g1[7] = 0u;

    // stage tile for key block kt into buffer bufi
    auto stage = [&](int bufi, int kt) {
        if (wid == 0) {
            unsigned long long ga = vp_addr + (unsigned long long)kt * 2ull;
            unsigned ga_lo = __builtin_amdgcn_readfirstlane((unsigned)ga);
            unsigned ga_hi = __builtin_amdgcn_readfirstlane((unsigned)(ga >> 32));
            u32x4 g0;
            g0[0] = 1u;                          // count=1
            g0[1] = bufi ? vb_lds1 : vb_lds0;    // lds_addr
            g0[2] = ga_lo;                       // global_addr[31:0]
            g0[3] = ga_hi | (2u << 30);          // global_addr[56:32] | type=2
            asm volatile("tensor_load_to_lds %0, %1" :: "s"(g0), "s"(g1) : "memory");
        }
        // contiguous K tile: 4KB = 256 x 16B chunks across 128 threads
#pragma unroll
        for (int c = 0; c < 2; ++c) {
            int boff = (t + c * 128) * 16;                 // byte offset in tile
            unsigned la = (unsigned)(unsigned long long)((const char*)kbuf[bufi] + boff);
            int goff = kt * (HD_ * 2) + boff;              // byte offset in k16
            asm volatile("global_load_async_to_lds_b128 %0, %1, %2"
                         :: "v"(la), "v"(goff), "s"(kp) : "memory");
        }
    };

    v16h qa0 = load_a_frag(qp, qbase, HD_, 0, lane);
    v16h qa1 = load_a_frag(qp, qbase, HD_, 32, lane);

    v8f acc[16] = {};
    float rmax[8], rsum[8];
    int qh[8], qw[8];
#pragma unroll
    for (int j = 0; j < 8; ++j) {
        rmax[j] = -3.0e38f; rsum[j] = 0.f;
        int qr = qbase + j + hi * 8;
        qh[j] = qr / W_; qw[j] = qr % W_;
    }
    __syncthreads();
    stage(0, 0);   // prologue: first tile in flight

    for (int kt = 0; kt < N_; kt += 32) {
        int cur = (kt >> 5) & 1;
        bool more = (kt + 32) < N_;
        if (more) stage(cur ^ 1, kt + 32);     // prefetch next tile (other buffer)
        // wait until *current* tile has landed (async/tensor ops complete in order)
        if (more) {
            asm volatile("s_wait_asynccnt 2" ::: "memory");
            if (wid == 0) __builtin_amdgcn_s_wait_tensorcnt(1);
        } else {
            asm volatile("s_wait_asynccnt 0" ::: "memory");
            if (wid == 0) __builtin_amdgcn_s_wait_tensorcnt(0);
        }
        __syncthreads();                        // tile visible to all waves
        const half_t* kb = kbuf[cur];
        const half_t* vb = vbuf[cur];

        // ---- scores: S = Q . K^T (from LDS) ----
        v8f s0 = {}, s1 = {};
        s0 = wmma_f16(qa0, load_b_frag(kb, 0, HD_, 0, lane), s0);
        s0 = wmma_f16(qa1, load_b_frag(kb, 0, HD_, 32, lane), s0);
        s1 = wmma_f16(qa0, load_b_frag(kb, 16, HD_, 0, lane), s1);
        s1 = wmma_f16(qa1, load_b_frag(kb, 16, HD_, 32, lane), s1);

        int key0 = kt + n16, key1 = key0 + 16;
        int kh0 = key0 / W_, kw0 = key0 % W_;
        int kh1 = key1 / W_, kw1 = key1 % W_;
#pragma unroll
        for (int j = 0; j < 8; ++j) {
            int dh0 = kh0 - qh[j] + RR_, dw0 = kw0 - qw[j] + RR_;
            int dh1 = kh1 - qh[j] + RR_, dw1 = kw1 - qw[j] + RR_;
            float b0 = ((unsigned)dh0 < 15u && (unsigned)dw0 < 15u) ? tbl[dh0 * 15 + dw0] : 0.f;
            float b1 = ((unsigned)dh1 < 15u && (unsigned)dw1 < 15u) ? tbl[dh1 * 15 + dw1] : 0.f;
            float x0 = fminf(fmaxf(s0[j] * 0.125f, -500.f), 500.f) + b0;
            float x1 = fminf(fmaxf(s1[j] * 0.125f, -500.f), 500.f) + b1;
            float tm = redmax16(fmaxf(x0, x1));
            float nm = fmaxf(rmax[j], tm);
            float sc = __expf(rmax[j] - nm);
            rmax[j] = nm;
            float p0 = __expf(x0 - nm), p1 = __expf(x1 - nm);
            rsum[j] = rsum[j] * sc + redsum16(p0 + p1);
#pragma unroll
            for (int ft = 0; ft < 16; ++ft) acc[ft][j] *= sc;
            pb[(j + hi * 8) * 32 + n16]      = (half_t)p0;
            pb[(j + hi * 8) * 32 + 16 + n16] = (half_t)p1;
        }
        asm volatile("s_wait_dscnt 0" ::: "memory");
        v16h pa = load_a_frag(pb, 0, 32, 0, lane);
#pragma unroll
        for (int ft = 0; ft < 16; ++ft)
            acc[ft] = wmma_f16(pa, load_b_frag(vb, ft * 16, 32, 0, lane), acc[ft]);
        __syncthreads();   // tile consumed; its buffer may be restaged next iter
    }
#pragma unroll
    for (int j = 0; j < 8; ++j) {
        float inv = 1.f / rsum[j];
        int row = qbase + j + hi * 8;
        half_t* op = fusion16 + (bm * N_ + row) * F_;
#pragma unroll
        for (int ft = 0; ft < 16; ++ft)
            op[ft * 16 + n16] = (half_t)(acc[ft][j] * inv);
    }
}

// ---------------- mid = gelu(fusion @ Wmid^T + bmid) ----------------
__global__ __launch_bounds__(128)
void mid_gemm_kernel(const half_t* __restrict__ fusion16, const half_t* __restrict__ wmid,
                     const float* __restrict__ bmid, half_t* __restrict__ mid16) {
    int lane = threadIdx.x, wid = threadIdx.y;
    int rowbase = blockIdx.x * 16;
    int co = blockIdx.y * 64 + wid * 16;
    v8f acc = {};
    for (int k = 0; k < F_; k += 32) {
        v16h a = load_a_frag(fusion16, rowbase, F_, k, lane);
        v16h bf = load_b_frag(wmid, co, F_, k, lane);
        acc = wmma_f16(a, bf, acc);
    }
    int n16 = lane & 15, hi = lane >> 4;
    int col = co + n16;
    float bias = bmid[col];
#pragma unroll
    for (int j = 0; j < 8; ++j) {
        int row = rowbase + j + hi * 8;
        float xv = acc[j] + bias;
        float g = 0.5f * xv * (1.f + erff(xv * 0.70710678f));
        mid16[(size_t)row * F_ + col] = (half_t)g;
    }
}

// ---------------- out = LN(mid@Wout^T + bout + fusion) ; mode score ----------------
__global__ __launch_bounds__(128)
void out_ln_kernel(const half_t* __restrict__ mid16, const half_t* __restrict__ wout,
                   const float* __restrict__ bout, const half_t* __restrict__ fusion16,
                   const float* __restrict__ lng, const float* __restrict__ lnb,
                   const float* __restrict__ wsc, const float* __restrict__ bsc,
                   float* __restrict__ out32, float* __restrict__ scores) {
    int lane = threadIdx.x, wid = threadIdx.y;
    int rowbase = (blockIdx.x * 4 + wid) * 16;
    v8f acc[16] = {};
    for (int k = 0; k < F_; k += 32) {
        v16h a = load_a_frag(mid16, rowbase, F_, k, lane);
#pragma unroll
        for (int ft = 0; ft < 16; ++ft)
            acc[ft] = wmma_f16(a, load_b_frag(wout, ft * 16, F_, k, lane), acc[ft]);
    }
    int n16 = lane & 15, hi = lane >> 4;
    float gcol[16], bcol[16], wcol[16], bocol[16];
#pragma unroll
    for (int ft = 0; ft < 16; ++ft) {
        int c = ft * 16 + n16;
        gcol[ft] = lng[c]; bcol[ft] = lnb[c]; wcol[ft] = wsc[c]; bocol[ft] = bout[c];
    }
#pragma unroll
    for (int j = 0; j < 8; ++j) {
        int row = rowbase + j + hi * 8;
        const half_t* fp = fusion16 + (size_t)row * F_;
        float x[16];
        float s = 0.f;
#pragma unroll
        for (int ft = 0; ft < 16; ++ft) {
            x[ft] = acc[ft][j] + bocol[ft] + (float)fp[ft * 16 + n16];
            s += x[ft];
        }
        s = redsum16(s);
        float mu = s * (1.f / F_);
        float q = 0.f;
#pragma unroll
        for (int ft = 0; ft < 16; ++ft) { float d = x[ft] - mu; q += d * d; }
        q = redsum16(q);
        float inv = rsqrtf(q * (1.f / F_) + 1e-12f);
        float sc = 0.f;
        float* op = out32 + (size_t)row * F_;
#pragma unroll
        for (int ft = 0; ft < 16; ++ft) {
            float y = (x[ft] - mu) * inv * gcol[ft] + bcol[ft];
            op[ft * 16 + n16] = y;
            sc += y * wcol[ft];
        }
        sc = redsum16(sc) + bsc[0];
        if (n16 == 0) scores[row] = sc;   // lane0 -> row j, lane16 -> row j+8
    }
}

// ---------------- head-fusion combine + transpose to (B,C,H,W) ----------------
__global__ void combine_kernel(const float* __restrict__ out32, const float* __restrict__ scores,
                               float* __restrict__ dout) {
    size_t idx = (size_t)blockIdx.x * 256 + threadIdx.x;
    if (idx >= (size_t)B_ * N_ * F_) return;
    int f = idx & 255;
    size_t nn = idx >> 8;
    int n = (int)(nn % N_);
    int b = (int)(nn / N_);
    float s[M_], mx = -3.0e38f;
#pragma unroll
    for (int m = 0; m < M_; ++m) { s[m] = scores[((size_t)(b * M_ + m) * N_) + n]; mx = fmaxf(mx, s[m]); }
    float e[M_], den = 0.f;
#pragma unroll
    for (int m = 0; m < M_; ++m) { e[m] = __expf(s[m] - mx); den += e[m]; }
    float o = 0.f;
#pragma unroll
    for (int m = 0; m < M_; ++m)
        o += out32[(((size_t)(b * M_ + m) * N_) + n) * F_ + f] * e[m];
    dout[((size_t)b * C_ + f) * N_ + n] = o / den;
}

extern "C" void kernel_launch(void* const* d_in, const int* in_sizes, int n_in,
                              void* d_out, int out_size, void* d_ws, size_t ws_size,
                              hipStream_t stream) {
    (void)in_sizes; (void)n_in; (void)out_size; (void)ws_size;
    const float* x    = (const float*)d_in[0];
    const float* Wq   = (const float*)d_in[1];
    const float* bq   = (const float*)d_in[2];
    const float* Wk   = (const float*)d_in[3];
    const float* bk   = (const float*)d_in[4];
    const float* Wv   = (const float*)d_in[5];
    const float* ptab = (const float*)d_in[6];
    const float* Wmid = (const float*)d_in[7];
    const float* bmid = (const float*)d_in[8];
    const float* Wout = (const float*)d_in[9];
    const float* bout = (const float*)d_in[10];
    const float* lng  = (const float*)d_in[11];
    const float* lnb  = (const float*)d_in[12];
    const float* wsc  = (const float*)d_in[13];
    const float* bsc  = (const float*)d_in[14];
    float* out = (float*)d_out;

    char* ws = (char*)d_ws;
    size_t off = 0;
    auto take = [&](size_t bytes) -> char* {
        char* p = ws + off;
        off += (bytes + 255) & ~(size_t)255;
        return p;
    };
    half_t* feat16   = (half_t*)take((size_t)B_ * N_ * C_ * 2);
    half_t* wq16     = (half_t*)take((size_t)C_ * C_ * 2);
    half_t* wk16     = (half_t*)take((size_t)C_ * C_ * 2);
    half_t* wv16     = (half_t*)take((size_t)M_ * F_ * C_ * 2);
    half_t* wmid16   = (half_t*)take((size_t)F_ * F_ * 2);
    half_t* wout16   = (half_t*)take((size_t)F_ * F_ * 2);
    half_t* q16      = (half_t*)take((size_t)B_ * M_ * N_ * HD_ * 2);
    half_t* k16      = (half_t*)take((size_t)B_ * M_ * N_ * HD_ * 2);
    half_t* vt16     = (half_t*)take((size_t)B_ * M_ * F_ * N_ * 2);
    half_t* fusion16 = (half_t*)take((size_t)B_ * M_ * N_ * F_ * 2);
    half_t* mid16    = (half_t*)take((size_t)B_ * M_ * N_ * F_ * 2);
    float*  out32    = (float*)take((size_t)B_ * M_ * N_ * F_ * 4);
    float*  scores   = (float*)take((size_t)B_ * M_ * N_ * 4);

    f2h_kernel<<<(C_ * C_ + 255) / 256, 256, 0, stream>>>(Wq, wq16, C_ * C_);
    f2h_kernel<<<(C_ * C_ + 255) / 256, 256, 0, stream>>>(Wk, wk16, C_ * C_);
    f2h_kernel<<<(M_ * F_ * C_ + 255) / 256, 256, 0, stream>>>(Wv, wv16, M_ * F_ * C_);
    f2h_kernel<<<(F_ * F_ + 255) / 256, 256, 0, stream>>>(Wmid, wmid16, F_ * F_);
    f2h_kernel<<<(F_ * F_ + 255) / 256, 256, 0, stream>>>(Wout, wout16, F_ * F_);

    ln_feat_kernel<<<dim3(B_ * N_ / 8), dim3(32, 8), 0, stream>>>(x, feat16);

    qkv_gemm_kernel<<<dim3(B_ * N_ / 16, (C_ * 2 + M_ * F_) / 64), dim3(32, 4), 0, stream>>>(
        feat16, wq16, wk16, wv16, bq, bk, q16, k16, vt16);

    attn_kernel<<<dim3(N_ / 64, M_, B_), dim3(32, 4), 0, stream>>>(
        q16, k16, vt16, ptab, fusion16);

    mid_gemm_kernel<<<dim3(B_ * M_ * N_ / 16, F_ / 64), dim3(32, 4), 0, stream>>>(
        fusion16, wmid16, bmid, mid16);

    out_ln_kernel<<<dim3(B_ * M_ * N_ / 64), dim3(32, 4), 0, stream>>>(
        mid16, wout16, bout, fusion16, lng, lnb, wsc, bsc, out32, scores);

    combine_kernel<<<dim3(B_ * N_ * F_ / 256), 256, 0, stream>>>(out32, scores, out);
}
